// LayerNormLSTMCell_55508157333862
// MI455X (gfx1250) — compile-verified, matrix operand order checked by reading, not run on hardware
//
#include <hip/hip_runtime.h>
#include <hip/hip_bf16.h>

// Shapes fixed by the reference: B=4096, I=H=1024, K=I+H=2048, N=4H=4096.
constexpr int Bsz = 4096;
constexpr int Hh  = 1024;
constexpr int Kd  = 2048;
constexpr int N4  = 4096;
constexpr float EPS = 1e-5f;

typedef __bf16 bf16_t;
typedef __attribute__((ext_vector_type(16))) __bf16 v16bf;
typedef __attribute__((ext_vector_type(8)))  float  v8f;

union ABu { v16bf v; uint4 u[2]; };

// ---------------------------------------------------------------------------
// Kernel 1: convert W -> bf16 (row major == B=W^T column-major) and build
// XH = [x | h] bf16 row-major [B, K]. float4 loads, 8B bf16x4 stores.
// ---------------------------------------------------------------------------
__global__ __launch_bounds__(256)
void pack_kernel(const float* __restrict__ x, const float* __restrict__ h,
                 const float* __restrict__ W,
                 bf16_t* __restrict__ XH, bf16_t* __restrict__ Wb) {
  const size_t i = (size_t)blockIdx.x * blockDim.x + threadIdx.x;  // 4-elem chunk
  constexpr size_t nchunks = (size_t)Bsz * Kd / 4;                 // 2,097,152
  if (i >= nchunks) return;

  const size_t r  = i / (Kd / 4);
  const size_t c4 = (i % (Kd / 4)) * 4;
  const float4 v = (c4 < (size_t)Hh) ? *(const float4*)(x + r * Hh + c4)
                                     : *(const float4*)(h + r * Hh + (c4 - Hh));
  union { bf16_t b[4]; uint2 u; } o;
  o.b[0] = (bf16_t)v.x; o.b[1] = (bf16_t)v.y;
  o.b[2] = (bf16_t)v.z; o.b[3] = (bf16_t)v.w;
  *(uint2*)(XH + i * 4) = o.u;

  const float4 w = *(const float4*)(W + i * 4);
  union { bf16_t b[4]; uint2 u; } ow;
  ow.b[0] = (bf16_t)w.x; ow.b[1] = (bf16_t)w.y;
  ow.b[2] = (bf16_t)w.z; ow.b[3] = (bf16_t)w.w;
  *(uint2*)(Wb + i * 4) = ow.u;
}

// ---------------------------------------------------------------------------
// Kernel 2: gates = XH @ Wb^T via v_wmma_f32_16x16x32_bf16.
// Block tile 128x128, K-step 64, double-buffered LDS filled by
// GLOBAL_LOAD_ASYNC_TO_LDS_B128 (ASYNCcnt). 8 waves (wave32) in 2(M)x4(N),
// each wave owns 64x32 = 4x2 fragments; 16 WMMAs per stage per wave.
// LDS: 16B-chunk XOR swizzle -> conflict-free ds_load_b128 fragment reads.
// ---------------------------------------------------------------------------
constexpr int BM = 128, BN = 128, BK = 64;
constexpr int CPR = BK / 8;  // 16B chunks per tile row = 8

__device__ __forceinline__ int swz(int row, int chunk) {
  return (row << 3) + (chunk ^ ((row + (row >> 3)) & 7));
}

__global__ __launch_bounds__(256)
void gemm_kernel(const bf16_t* __restrict__ XH, const bf16_t* __restrict__ Wb,
                 float* __restrict__ gates) {
  __shared__ uint4 As[2][BM * CPR];   // 2 * 128*8 * 16B = 32 KB
  __shared__ uint4 Bs[2][BN * CPR];   // 32 KB  (total exactly 64 KB)

  const int tid   = threadIdx.x;
  const int lane  = tid & 31;
  const int wave  = tid >> 5;
  const int waveM = wave >> 2;          // 0..1  -> 64 rows each
  const int waveN = wave & 3;           // 0..3  -> 32 cols each
  const int m0 = blockIdx.y * BM;
  const int n0 = blockIdx.x * BN;

  const int lrow = lane & 15;           // row/col within a 16x16 fragment
  const int hsel = lane >> 4;           // K-half select (ISA 7.12.2 layout)

  // Issue one stage of async global->LDS copies (8 x b128 per thread).
  auto issue_stage = [&](int buf, int kt) {
#pragma unroll
    for (int q = 0; q < 4; ++q) {
      const int f   = tid + (q << 8);   // 0..1023
      const int row = f >> 3;           // 0..127
      const int ch  = f & 7;            // 16B chunk within row
      const bf16_t* ga = XH + (size_t)(m0 + row) * Kd + kt + ch * 8;
      const bf16_t* gb = Wb + (size_t)(n0 + row) * Kd + kt + ch * 8;
      const uint32_t la = (uint32_t)(uintptr_t)&As[buf][swz(row, ch)];
      const uint32_t lb = (uint32_t)(uintptr_t)&Bs[buf][swz(row, ch)];
      asm volatile("global_load_async_to_lds_b128 %0, %1, off"
                   :: "v"(la), "v"(ga) : "memory");
      asm volatile("global_load_async_to_lds_b128 %0, %1, off"
                   :: "v"(lb), "v"(gb) : "memory");
    }
  };

  v8f acc[4][2] = {};

  issue_stage(0, 0);                           // prime the pipeline
  constexpr int NSTAGE = Kd / BK;              // 32
  for (int s = 0; s < NSTAGE; ++s) {
    // Drain this wave's outstanding async copies, then rendezvous: after the
    // barrier, buf[s&1] is fully written by all waves and buf[(s+1)&1] is no
    // longer being read by anyone (its last reads were in stage s-1).
    asm volatile("s_wait_asynccnt 0x0" ::: "memory");
    __syncthreads();
    if (s + 1 < NSTAGE) issue_stage((s + 1) & 1, (s + 1) * BK);

    const int buf = s & 1;
#pragma unroll
    for (int ks = 0; ks < 2; ++ks) {           // two 16x16x32 K-steps per stage
      const int cb = ks * 4 + hsel;            // chunk of K khalf..khalf+7
      ABu af[4];
      ABu bfr[2];
#pragma unroll
      for (int mf = 0; mf < 4; ++mf) {
        const int r = waveM * 64 + mf * 16 + lrow;
        af[mf].u[0] = As[buf][swz(r, cb)];     // K khalf .. +7
        af[mf].u[1] = As[buf][swz(r, cb + 2)]; // K khalf+16 .. +23
      }
#pragma unroll
      for (int nf = 0; nf < 2; ++nf) {
        const int r = waveN * 32 + nf * 16 + lrow;
        bfr[nf].u[0] = Bs[buf][swz(r, cb)];
        bfr[nf].u[1] = Bs[buf][swz(r, cb + 2)];
      }
#pragma unroll
      for (int mf = 0; mf < 4; ++mf)
#pragma unroll
        for (int nf = 0; nf < 2; ++nf)
          acc[mf][nf] = __builtin_amdgcn_wmma_f32_16x16x32_bf16(
              false, af[mf].v, false, bfr[nf].v,
              (short)0, acc[mf][nf], false, false);
    }
  }

  // Store C (f32). D layout: VGPR v -> M=v (lanes 0-15) / M=v+8 (lanes 16-31).
  const int rsel = hsel * 8;
#pragma unroll
  for (int mf = 0; mf < 4; ++mf) {
#pragma unroll
    for (int nf = 0; nf < 2; ++nf) {
      const int col = n0 + waveN * 32 + nf * 16 + lrow;
#pragma unroll
      for (int v = 0; v < 8; ++v) {
        const int row = m0 + waveM * 64 + mf * 16 + rsel + v;
        gates[(size_t)row * N4 + col] = acc[mf][nf][v];
      }
    }
  }
}

// ---------------------------------------------------------------------------
// Kernel 3: fused bias + 4 gate LayerNorms + activations + cell update +
// LN(c_new) + h_new. One block (256 threads, 8 wave32) per batch row.
// ---------------------------------------------------------------------------
__device__ __forceinline__ float block_reduce_sum(float v, float* sbuf) {
#pragma unroll
  for (int off = 16; off > 0; off >>= 1) v += __shfl_xor(v, off, 32);
  const int lane = threadIdx.x & 31;
  const int wid  = threadIdx.x >> 5;
  if (lane == 0) sbuf[wid] = v;
  __syncthreads();
  float r = (threadIdx.x < 8) ? sbuf[threadIdx.x] : 0.0f;
  if (wid == 0) {
#pragma unroll
    for (int off = 4; off > 0; off >>= 1) r += __shfl_xor(r, off, 32);
    if (lane == 0) sbuf[0] = r;
  }
  __syncthreads();
  const float out = sbuf[0];
  __syncthreads();
  return out;
}

__device__ __forceinline__ float sigmoidf(float x) {
  return 1.0f / (1.0f + __expf(-x));
}

__global__ __launch_bounds__(256)
void epilogue_kernel(const float* __restrict__ gates, const float* __restrict__ c,
                     const float* __restrict__ bias,
                     const float* __restrict__ lnig, const float* __restrict__ lnib,
                     const float* __restrict__ lnfg, const float* __restrict__ lnfb,
                     const float* __restrict__ lngg, const float* __restrict__ lngb,
                     const float* __restrict__ lnog, const float* __restrict__ lnob,
                     const float* __restrict__ lncg, const float* __restrict__ lncb,
                     float* __restrict__ out) {
  __shared__ float sred[32];
  const int b = blockIdx.x;
  const int t = threadIdx.x;
  const float invH = 1.0f / (float)Hh;
  const size_t rowBase = (size_t)b * N4;

  float iv[4], fv[4], gv[4], ov[4];
  float s_i = 0, q_i = 0, s_f = 0, q_f = 0, s_g = 0, q_g = 0, s_o = 0, q_o = 0;
#pragma unroll
  for (int e = 0; e < 4; ++e) {
    const int j = t + e * 256;
    const float a  = gates[rowBase + 0 * Hh + j] + bias[0 * Hh + j];
    const float f_ = gates[rowBase + 1 * Hh + j] + bias[1 * Hh + j];
    const float g_ = gates[rowBase + 2 * Hh + j] + bias[2 * Hh + j];
    const float o_ = gates[rowBase + 3 * Hh + j] + bias[3 * Hh + j];
    iv[e] = a;  fv[e] = f_; gv[e] = g_; ov[e] = o_;
    s_i += a;  q_i += a * a;   s_f += f_; q_f += f_ * f_;
    s_g += g_; q_g += g_ * g_; s_o += o_; q_o += o_ * o_;
  }
  s_i = block_reduce_sum(s_i, sred); q_i = block_reduce_sum(q_i, sred);
  s_f = block_reduce_sum(s_f, sred); q_f = block_reduce_sum(q_f, sred);
  s_g = block_reduce_sum(s_g, sred); q_g = block_reduce_sum(q_g, sred);
  s_o = block_reduce_sum(s_o, sred); q_o = block_reduce_sum(q_o, sred);

  const float mi = s_i * invH, ri = rsqrtf(q_i * invH - mi * mi + EPS);
  const float mf = s_f * invH, rf = rsqrtf(q_f * invH - mf * mf + EPS);
  const float mg = s_g * invH, rg = rsqrtf(q_g * invH - mg * mg + EPS);
  const float mo = s_o * invH, ro = rsqrtf(q_o * invH - mo * mo + EPS);

  float cn[4];
  float s_c = 0, q_c = 0;
#pragma unroll
  for (int e = 0; e < 4; ++e) {
    const int j = t + e * 256;
    const float ig = sigmoidf((iv[e] - mi) * ri * lnig[j] + lnib[j]);
    const float fg = sigmoidf((fv[e] - mf) * rf * lnfg[j] + lnfb[j]);
    const float gg = tanhf(   (gv[e] - mg) * rg * lngg[j] + lngb[j]);
    const float og = sigmoidf((ov[e] - mo) * ro * lnog[j] + lnob[j]);
    const float cnew = fg * c[(size_t)b * Hh + j] + ig * gg;
    cn[e] = cnew;
    ov[e] = og;                       // keep output gate
    s_c += cnew; q_c += cnew * cnew;
  }
  s_c = block_reduce_sum(s_c, sred);
  q_c = block_reduce_sum(q_c, sred);
  const float mc = s_c * invH, rc = rsqrtf(q_c * invH - mc * mc + EPS);

#pragma unroll
  for (int e = 0; e < 4; ++e) {
    const int j = t + e * 256;
    const float lnc  = (cn[e] - mc) * rc * lncg[j] + lncb[j];
    const float hnew = ov[e] * tanhf(lnc);
    out[(size_t)b * Hh + j]                    = hnew;   // h_new first
    out[(size_t)Bsz * Hh + (size_t)b * Hh + j] = cn[e];  // then c_new
  }
}

// ---------------------------------------------------------------------------
// Launch. Workspace layout (needs 96 MB):
//   [0, 16MB)   Wb  bf16 [4096, 2048]
//   [16, 32MB)  XH  bf16 [4096, 2048]
//   [32, 96MB)  gates f32 [4096, 4096]
// ---------------------------------------------------------------------------
extern "C" void kernel_launch(void* const* d_in, const int* in_sizes, int n_in,
                              void* d_out, int out_size, void* d_ws, size_t ws_size,
                              hipStream_t stream) {
  const float* x    = (const float*)d_in[0];
  const float* h    = (const float*)d_in[1];
  const float* c    = (const float*)d_in[2];
  const float* W    = (const float*)d_in[3];
  const float* bias = (const float*)d_in[4];
  const float* lnig = (const float*)d_in[5];
  const float* lnib = (const float*)d_in[6];
  const float* lnfg = (const float*)d_in[7];
  const float* lnfb = (const float*)d_in[8];
  const float* lngg = (const float*)d_in[9];
  const float* lngb = (const float*)d_in[10];
  const float* lnog = (const float*)d_in[11];
  const float* lnob = (const float*)d_in[12];
  const float* lncg = (const float*)d_in[13];
  const float* lncb = (const float*)d_in[14];
  float* out = (float*)d_out;

  bf16_t* Wb    = (bf16_t*)d_ws;
  bf16_t* XH    = Wb + (size_t)N4 * Kd;
  float*  gates = (float*)(XH + (size_t)Bsz * Kd);

  const size_t nchunks = (size_t)Bsz * Kd / 4;          // 2,097,152
  const int packBlocks = (int)((nchunks + 255) / 256);  // 8192
  pack_kernel<<<packBlocks, 256, 0, stream>>>(x, h, W, XH, Wb);

  dim3 gemmGrid(N4 / BN, Bsz / BM);                     // 32 x 32 blocks
  gemm_kernel<<<gemmGrid, 256, 0, stream>>>(XH, Wb, gates);

  epilogue_kernel<<<Bsz, 256, 0, stream>>>(gates, c, bias,
                                           lnig, lnib, lnfg, lnfb,
                                           lngg, lngb, lnog, lnob,
                                           lncg, lncb, out);
}
